// EncoderLayer_15375982920173
// MI455X (gfx1250) — compile-verified, matrix operand order checked by reading, not run on hardware
//
#include <hip/hip_runtime.h>

// ---------------------------------------------------------------------------
// CDNA5 (gfx1250) transformer encoder layer
//   - v_wmma_f32_16x16x32_bf16 for all GEMMs (fp32 accum)
//   - global_load_async_to_lds_b128 (inline asm, ASYNCcnt) double-buffered
//     staging: the clang builtin exists but takes OpenCL-AS pointers that HIP
//     cannot spell, so we emit the instruction directly.
//   - weights pre-transposed so A and B^T tiles are both K-contiguous
// ---------------------------------------------------------------------------

typedef __attribute__((ext_vector_type(16))) __bf16 v16bf;
typedef __attribute__((ext_vector_type(8)))  __bf16 bf16x8;
typedef __attribute__((ext_vector_type(8)))  float  v8f;

#define B_   4
#define T_   1024
#define C_   1024
#define H_   16
#define HS_  64
#define DFF_ 4096
#define M_ROWS (B_ * T_)   // 4096

__device__ __forceinline__ v8f wmma_bf16(v16bf a, v16bf b, v8f c) {
  return __builtin_amdgcn_wmma_f32_16x16x32_bf16(
      false, a, false, b, (short)0, c, false, false);
}

// ---------------------------------------------------------------------------
// Async global->LDS copy (16B per lane) + ASYNCcnt waits, via inline asm.
// Generic pointers to __shared__ have the LDS byte offset in addr[31:0]
// (ISA: LDS aperture address mapping), so truncation yields the VDST value.
// ---------------------------------------------------------------------------
__device__ __forceinline__ void copy16_g2l(const void* g, void* l) {
#if defined(__AMDGCN__)
  const unsigned lds = (unsigned)(unsigned long long)l;
  asm volatile("global_load_async_to_lds_b128 %0, %1, off"
               :: "v"(lds), "v"(g) : "memory");
#else
  *reinterpret_cast<bf16x8*>(l) = *reinterpret_cast<const bf16x8*>(g);
#endif
}

__device__ __forceinline__ void async_wait_le4() {
#if defined(__AMDGCN__)
  asm volatile("s_wait_asynccnt 0x4" ::: "memory");
#endif
}
__device__ __forceinline__ void async_wait_le0() {
#if defined(__AMDGCN__)
  asm volatile("s_wait_asynccnt 0x0" ::: "memory");
#endif
}

// ---------------------------------------------------------------------------
// Fragment loaders.
// 16-bit A 16x32 per-lane layout: lanes 0-15 -> M=lane, K {0..7,16..23};
//                                 lanes 16-31 -> M=lane-16, K {8..15,24..31}.
// B-fragments use the same layout on a [N][K] (transposed) view.
// pitch*2 bytes must be 16B-aligned.
// ---------------------------------------------------------------------------
__device__ __forceinline__ v16bf frag_lds(const __bf16* base, int row, int pitch, int koff) {
  const int lane = threadIdx.x & 31;
  const int r  = row + (lane & 15);
  const int kb = ((lane < 16) ? 0 : 8) + koff;
  const bf16x8 c0 = *reinterpret_cast<const bf16x8*>(base + (size_t)r * pitch + kb);
  const bf16x8 c1 = *reinterpret_cast<const bf16x8*>(base + (size_t)r * pitch + kb + 16);
  v16bf f;
#pragma unroll
  for (int i = 0; i < 8; ++i) { f[i] = c0[i]; f[8 + i] = c1[i]; }
  return f;
}

__device__ __forceinline__ v16bf frag_global(const __bf16* p, size_t stride, int koff) {
  const int lane = threadIdx.x & 31;
  const int r  = lane & 15;
  const int kb = ((lane < 16) ? 0 : 8) + koff;
  const bf16x8 c0 = *reinterpret_cast<const bf16x8*>(p + (size_t)r * stride + kb);
  const bf16x8 c1 = *reinterpret_cast<const bf16x8*>(p + (size_t)r * stride + kb + 16);
  v16bf f;
#pragma unroll
  for (int i = 0; i < 8; ++i) { f[i] = c0[i]; f[8 + i] = c1[i]; }
  return f;
}

// ---------------------------------------------------------------------------
// LayerNorm: one block per row (C=1024), fp32 in, bf16 out.
// ---------------------------------------------------------------------------
__global__ __launch_bounds__(256)
void ln_kernel(const float* __restrict__ x, const float* __restrict__ g,
               const float* __restrict__ be, __bf16* __restrict__ out) {
  const int row  = blockIdx.x;
  const int tid  = threadIdx.x;
  const int wave = tid >> 5, lane = tid & 31;
  const float* xr = x + (size_t)row * C_;

  float s = 0.f, s2 = 0.f;
#pragma unroll
  for (int c = tid; c < C_; c += 256) { float v = xr[c]; s += v; s2 += v * v; }
#pragma unroll
  for (int o = 16; o >= 1; o >>= 1) {
    s  += __shfl_xor(s,  o, 32);
    s2 += __shfl_xor(s2, o, 32);
  }
  __shared__ float sh[18];
  if (lane == 0) { sh[wave] = s; sh[8 + wave] = s2; }
  __syncthreads();
  if (tid == 0) {
    float a = 0.f, b2 = 0.f;
    for (int w = 0; w < 8; ++w) { a += sh[w]; b2 += sh[8 + w]; }
    sh[16] = a; sh[17] = b2;
  }
  __syncthreads();
  const float mu   = sh[16] * (1.0f / C_);
  const float var  = sh[17] * (1.0f / C_) - mu * mu;
  const float rstd = rsqrtf(var + 1e-5f);
  for (int c = tid; c < C_; c += 256)
    out[(size_t)row * C_ + c] = (__bf16)((xr[c] - mu) * rstd * g[c] + be[c]);
}

// ---------------------------------------------------------------------------
// Weight converters (all produce transposed [N][K] bf16 for K-contiguous B)
// ---------------------------------------------------------------------------
// src row-major [K][N] fp32 -> dst [N][K] bf16
__global__ __launch_bounds__(256)
void cvt_t_kernel(const float* __restrict__ src, __bf16* __restrict__ dst,
                  int K, int N) {
  int i = blockIdx.x * 256 + threadIdx.x;   // i < K*N
  int k = i / N, n = i - k * N;
  dst[(size_t)n * K + k] = (__bf16)src[i];
}

// wq (H, C, HS) fp32 -> [H*HS][C] bf16   (row n = h*HS+d, contiguous in c)
__global__ __launch_bounds__(256)
void repack_qkv_t_kernel(const float* __restrict__ w, __bf16* __restrict__ out) {
  int i = blockIdx.x * 256 + threadIdx.x;   // i < H*C*HS
  int h = i / (C_ * HS_);
  int rem = i - h * (C_ * HS_);
  int c = rem / HS_;
  int d = rem - c * HS_;
  out[(size_t)(h * HS_ + d) * C_ + c] = (__bf16)w[i];
}

// ---------------------------------------------------------------------------
// GEMM: C[M,N] = A[M,K] @ Bt[N,K]^T  (+bias, relu, residual)
// 256 threads / 8 waves; block tile 128x128; wave tile 32x64 (2x4 WMMA);
// K staged in 32-wide slices, double-buffered async global->LDS copies.
// OUTMODE: 0 = fp32 row-major, 1 = bf16 row-major, 2 = bf16 transposed [N][M]
// ---------------------------------------------------------------------------
template <bool BIAS, bool RELU, bool RESID, int OUTMODE>
__global__ __launch_bounds__(256)
void gemm_abt_kernel(const __bf16* __restrict__ A, const __bf16* __restrict__ Bt,
                     const float* __restrict__ bias, const float* __restrict__ resid,
                     void* __restrict__ out, int M, int N, int K) {
  __shared__ __bf16 As[2][128][40];   // pitch 40 elems = 80B (16B aligned)
  __shared__ __bf16 Bs[2][128][40];

  const int tid  = threadIdx.x;
  const int wave = tid >> 5, lane = tid & 31;
  const int m0 = blockIdx.y * 128;
  const int n0 = blockIdx.x * 128;
  const int wm = (wave & 3) * 32;    // 0,32,64,96
  const int wn = (wave >> 2) * 64;   // 0,64

  v8f acc[2][4] = {};

  auto stage = [&](int buf, int k0) {
#pragma unroll
    for (int it = 0; it < 2; ++it) {
      const int idx = (tid + it * 256) * 8;   // 0..4088
      const int r = idx >> 5, c = idx & 31;
      copy16_g2l(A + (size_t)(m0 + r) * K + k0 + c, &As[buf][r][c]);
    }
#pragma unroll
    for (int it = 0; it < 2; ++it) {
      const int idx = (tid + it * 256) * 8;
      const int r = idx >> 5, c = idx & 31;
      copy16_g2l(Bt + (size_t)(n0 + r) * K + k0 + c, &Bs[buf][r][c]);
    }
  };

  stage(0, 0);
  int buf = 0;
  for (int k0 = 0; k0 < K; k0 += 32) {
    if (k0 + 32 < K) {
      stage(buf ^ 1, k0 + 32);   // 4 new async copies per thread
      async_wait_le4();          // retire the older batch (in-order completion)
    } else {
      async_wait_le0();
    }
    __syncthreads();

    const v16bf a0 = frag_lds(&As[buf][0][0], wm,      40, 0);
    const v16bf a1 = frag_lds(&As[buf][0][0], wm + 16, 40, 0);
#pragma unroll
    for (int j = 0; j < 4; ++j) {
      const v16bf bj = frag_lds(&Bs[buf][0][0], wn + j * 16, 40, 0);
      acc[0][j] = wmma_bf16(a0, bj, acc[0][j]);
      acc[1][j] = wmma_bf16(a1, bj, acc[1][j]);
    }
    __syncthreads();
    buf ^= 1;
  }

  // epilogue; C layout: lanes 0-15: M=r, N=lane ; lanes 16-31: M=r+8, N=lane-16
  const int rowoff = (lane < 16) ? 0 : 8;
#pragma unroll
  for (int i = 0; i < 2; ++i)
#pragma unroll
    for (int j = 0; j < 4; ++j) {
      const int col = n0 + wn + j * 16 + (lane & 15);
#pragma unroll
      for (int r = 0; r < 8; ++r) {
        const int row = m0 + wm + i * 16 + rowoff + r;
        float v = acc[i][j][r];
        if (BIAS)  v += bias[col];
        if (RELU)  v = fmaxf(v, 0.f);
        if (RESID) v += resid[(size_t)row * N + col];
        if (OUTMODE == 0)      ((float*)out)[(size_t)row * N + col] = v;
        else if (OUTMODE == 1) ((__bf16*)out)[(size_t)row * N + col] = (__bf16)v;
        else                   ((__bf16*)out)[(size_t)col * M + row] = (__bf16)v;
      }
    }
}

// ---------------------------------------------------------------------------
// Attention scores + softmax: one block per (b, h, 16-row tile of T).
// S strip (16 x 1024) entirely in registers (8 v8f per wave); K-fragments
// loaded straight from global (rows are K-contiguous bf16).
// ---------------------------------------------------------------------------
__global__ __launch_bounds__(256)
void attn_scores_kernel(const __bf16* __restrict__ q_bf, const __bf16* __restrict__ k_bf,
                        float* __restrict__ wei) {
  const int t0 = blockIdx.x * 16;
  const int h  = blockIdx.y;
  const int b  = blockIdx.z;
  const int tid  = threadIdx.x;
  const int wave = tid >> 5, lane = tid & 31;

  __shared__ __bf16 qt[16][64];
  __shared__ float  red[8][16];
  __shared__ float  rowmax[16];
  __shared__ float  rowsum[16];

  {
    int idx = tid * 4;
    int r = idx >> 6, c = idx & 63;
#pragma unroll
    for (int j = 0; j < 4; ++j)
      qt[r][c + j] = q_bf[((size_t)(b * T_) + t0 + r) * C_ + h * HS_ + c + j];
  }
  __syncthreads();

  const v16bf qa0 = frag_lds(&qt[0][0], 0, 64, 0);
  const v16bf qa1 = frag_lds(&qt[0][0], 0, 64, 32);

  v8f acc[8] = {};
  const __bf16* kbase = k_bf + ((size_t)(b * T_)) * C_ + h * HS_;
#pragma unroll
  for (int j = 0; j < 8; ++j) {
    const int s0 = wave * 128 + j * 16;
    const __bf16* kp = kbase + (size_t)s0 * C_;
    acc[j] = wmma_bf16(qa0, frag_global(kp, C_, 0),  acc[j]);
    acc[j] = wmma_bf16(qa1, frag_global(kp, C_, 32), acc[j]);
  }
#pragma unroll
  for (int j = 0; j < 8; ++j)
#pragma unroll
    for (int r = 0; r < 8; ++r) acc[j][r] *= 0.125f;   // HS^-0.5

  const int rb = (lane < 16) ? 0 : 8;

  // row max
  float p[8];
#pragma unroll
  for (int r = 0; r < 8; ++r) {
    p[r] = acc[0][r];
#pragma unroll
    for (int j = 1; j < 8; ++j) p[r] = fmaxf(p[r], acc[j][r]);
  }
#pragma unroll
  for (int o = 8; o >= 1; o >>= 1)
#pragma unroll
    for (int r = 0; r < 8; ++r) p[r] = fmaxf(p[r], __shfl_xor(p[r], o, 32));
  if ((lane & 15) == 0)
#pragma unroll
    for (int r = 0; r < 8; ++r) red[wave][rb + r] = p[r];
  __syncthreads();
  if (tid < 16) {
    float m = red[0][tid];
    for (int w = 1; w < 8; ++w) m = fmaxf(m, red[w][tid]);
    rowmax[tid] = m;
  }
  __syncthreads();

  // exp & row sum
  float ps[8];
#pragma unroll
  for (int r = 0; r < 8; ++r) {
    const float mx = rowmax[rb + r];
    float s = 0.f;
#pragma unroll
    for (int j = 0; j < 8; ++j) {
      float e = __expf(acc[j][r] - mx);
      acc[j][r] = e;
      s += e;
    }
    ps[r] = s;
  }
#pragma unroll
  for (int o = 8; o >= 1; o >>= 1)
#pragma unroll
    for (int r = 0; r < 8; ++r) ps[r] += __shfl_xor(ps[r], o, 32);
  if ((lane & 15) == 0)
#pragma unroll
    for (int r = 0; r < 8; ++r) red[wave][rb + r] = ps[r];
  __syncthreads();
  if (tid < 16) {
    float s = 0.f;
    for (int w = 0; w < 8; ++w) s += red[w][tid];
    rowsum[tid] = s;
  }
  __syncthreads();

  float* wrow = wei + (((size_t)b * H_ + h) * T_ + t0) * T_;
#pragma unroll
  for (int r = 0; r < 8; ++r) {
    const int m = rb + r;
    const float inv = 1.f / rowsum[m];
#pragma unroll
    for (int j = 0; j < 8; ++j) {
      const int s = wave * 128 + j * 16 + (lane & 15);
      wrow[(size_t)m * T_ + s] = acc[j][r] * inv;
    }
  }
}

// ---------------------------------------------------------------------------
// PV: attn[b, t, h*64+d] = sum_s wei[b,h,t,s] * v[b, s, h*64+d]
// vT is [C][B*T] bf16 (token-contiguous), so B-fragments stream from global.
// Only the fp32 wei tile is LDS-staged (needs bf16 conversion).
// Block: 128 rows of t; 8 waves each own 16 rows x 64 cols (4 WMMA tiles).
// ---------------------------------------------------------------------------
__global__ __launch_bounds__(256)
void attn_pv_kernel(const float* __restrict__ wei, const __bf16* __restrict__ vT,
                    __bf16* __restrict__ attn_bf) {
  __shared__ __bf16 As[128][40];

  const int bh = blockIdx.y;                 // b*16 + h
  const int b  = bh >> 4, h = bh & 15;
  const int m0 = blockIdx.x * 128;
  const int tid  = threadIdx.x;
  const int wave = tid >> 5, lane = tid & 31;
  const int wm = wave * 16;

  const float*  Ap  = wei + (size_t)bh * T_ * T_;
  const __bf16* vTp = vT + (size_t)(h * HS_) * M_ROWS + b * T_;  // row d: +d*M_ROWS
  __bf16*       Op  = attn_bf + (size_t)(b * T_) * C_ + h * HS_;

  v8f acc[4] = {};

  for (int k0 = 0; k0 < T_; k0 += 32) {
    // stage wei tile 128x32 (fp32 -> bf16)
#pragma unroll
    for (int it = 0; it < 2; ++it) {
      const int idx = (tid + it * 256) * 8;
      const int r = idx >> 5, c = idx & 31;
#pragma unroll
      for (int j = 0; j < 8; ++j)
        As[r][c + j] = (__bf16)Ap[(size_t)(m0 + r) * T_ + k0 + c + j];
    }
    __syncthreads();

    const v16bf a = frag_lds(&As[0][0], wm, 40, 0);
#pragma unroll
    for (int j = 0; j < 4; ++j) {
      const v16bf bj = frag_global(vTp + (size_t)(j * 16) * M_ROWS + k0, M_ROWS, 0);
      acc[j] = wmma_bf16(a, bj, acc[j]);
    }
    __syncthreads();
  }

  const int rowoff = (lane < 16) ? 0 : 8;
#pragma unroll
  for (int j = 0; j < 4; ++j) {
    const int col = j * 16 + (lane & 15);
#pragma unroll
    for (int r = 0; r < 8; ++r) {
      const int row = m0 + wm + rowoff + r;
      Op[(size_t)row * C_ + col] = (__bf16)acc[j][r];
    }
  }
}

// ---------------------------------------------------------------------------
// Host-side launch
// ---------------------------------------------------------------------------
extern "C" void kernel_launch(void* const* d_in, const int* in_sizes, int n_in,
                              void* d_out, int out_size, void* d_ws, size_t ws_size,
                              hipStream_t stream) {
  (void)in_sizes; (void)n_in; (void)out_size; (void)ws_size;

  const float* x      = (const float*)d_in[0];
  const float* wq     = (const float*)d_in[1];
  const float* wk     = (const float*)d_in[2];
  const float* wv     = (const float*)d_in[3];
  const float* w_proj = (const float*)d_in[4];
  const float* b_proj = (const float*)d_in[5];
  const float* w1     = (const float*)d_in[6];
  const float* b1     = (const float*)d_in[7];
  const float* w2     = (const float*)d_in[8];
  const float* b2     = (const float*)d_in[9];
  const float* g1     = (const float*)d_in[10];
  const float* be1    = (const float*)d_in[11];
  const float* g2     = (const float*)d_in[12];
  const float* be2    = (const float*)d_in[13];

  char* ws = (char*)d_ws;
  size_t off = 0;
  auto carve = [&](size_t bytes) -> void* {
    void* p = ws + off;
    off += (bytes + 255) & ~(size_t)255;
    return p;
  };
  __bf16* h_bf    = (__bf16*)carve((size_t)M_ROWS * C_ * 2);   // LN1 out
  __bf16* wq_t    = (__bf16*)carve((size_t)C_ * C_ * 2);       // [H*HS][C]
  __bf16* wk_t    = (__bf16*)carve((size_t)C_ * C_ * 2);
  __bf16* wv_t    = (__bf16*)carve((size_t)C_ * C_ * 2);
  __bf16* q_bf    = (__bf16*)carve((size_t)M_ROWS * C_ * 2);   // [B*T][C]
  __bf16* k_bf    = (__bf16*)carve((size_t)M_ROWS * C_ * 2);
  __bf16* vT_bf   = (__bf16*)carve((size_t)M_ROWS * C_ * 2);   // [C][B*T]
  __bf16* attn_bf = (__bf16*)carve((size_t)M_ROWS * C_ * 2);
  __bf16* wproj_t = (__bf16*)carve((size_t)C_ * C_ * 2);       // [C][C]
  float*  x2      = (float*) carve((size_t)M_ROWS * C_ * 4);
  __bf16* h2_bf   = (__bf16*)carve((size_t)M_ROWS * C_ * 2);
  __bf16* w1_t    = (__bf16*)carve((size_t)C_ * DFF_ * 2);     // [DFF][C]
  __bf16* w2_t    = (__bf16*)carve((size_t)DFF_ * C_ * 2);     // [C][DFF]
  __bf16* ff_bf   = (__bf16*)carve((size_t)M_ROWS * DFF_ * 2);

  float* out_x = (float*)d_out;
  float* wei   = (float*)d_out + (size_t)M_ROWS * C_;

  // 1) LN1
  ln_kernel<<<M_ROWS, 256, 0, stream>>>(x, g1, be1, h_bf);

  // 2) weight conversion (all transposed to [N][K])
  repack_qkv_t_kernel<<<(H_ * C_ * HS_) / 256, 256, 0, stream>>>(wq, wq_t);
  repack_qkv_t_kernel<<<(H_ * C_ * HS_) / 256, 256, 0, stream>>>(wk, wk_t);
  repack_qkv_t_kernel<<<(H_ * C_ * HS_) / 256, 256, 0, stream>>>(wv, wv_t);
  cvt_t_kernel<<<(C_ * C_) / 256, 256, 0, stream>>>(w_proj, wproj_t, C_, C_);
  cvt_t_kernel<<<(C_ * DFF_) / 256, 256, 0, stream>>>(w1, w1_t, C_, DFF_);
  cvt_t_kernel<<<(DFF_ * C_) / 256, 256, 0, stream>>>(w2, w2_t, DFF_, C_);

  // 3) QKV projections [4096,1024] @ [1024,1024]
  dim3 g1024(C_ / 128, M_ROWS / 128);
  gemm_abt_kernel<false, false, false, 1><<<g1024, 256, 0, stream>>>(
      h_bf, wq_t, nullptr, nullptr, q_bf, M_ROWS, C_, C_);
  gemm_abt_kernel<false, false, false, 1><<<g1024, 256, 0, stream>>>(
      h_bf, wk_t, nullptr, nullptr, k_bf, M_ROWS, C_, C_);
  gemm_abt_kernel<false, false, false, 2><<<g1024, 256, 0, stream>>>(   // transposed out
      h_bf, wv_t, nullptr, nullptr, vT_bf, M_ROWS, C_, C_);

  // 4) attention scores + softmax -> wei (fp32, second half of d_out)
  attn_scores_kernel<<<dim3(T_ / 16, H_, B_), 256, 0, stream>>>(q_bf, k_bf, wei);

  // 5) PV
  attn_pv_kernel<<<dim3(T_ / 128, B_ * H_), 256, 0, stream>>>(wei, vT_bf, attn_bf);

  // 6) output projection + bias + residual(x) -> x2 (fp32)
  gemm_abt_kernel<true, false, true, 0><<<g1024, 256, 0, stream>>>(
      attn_bf, wproj_t, b_proj, x, x2, M_ROWS, C_, C_);

  // 7) LN2
  ln_kernel<<<M_ROWS, 256, 0, stream>>>(x2, g2, be2, h2_bf);

  // 8) FFN up: relu(h2 @ w1 + b1) -> bf16 [4096,4096]
  gemm_abt_kernel<true, true, false, 1><<<dim3(DFF_ / 128, M_ROWS / 128), 256, 0, stream>>>(
      h2_bf, w1_t, b1, nullptr, ff_bf, M_ROWS, DFF_, C_);

  // 9) FFN down + bias + residual(x2) -> d_out (fp32)
  gemm_abt_kernel<true, false, true, 0><<<g1024, 256, 0, stream>>>(
      ff_bf, w2_t, b2, x2, out_x, M_ROWS, C_, DFF_);
}
